// LinformerAttention_59923383714149
// MI455X (gfx1250) — matrix operand order repaired and checked by reading, not compiled, not run on hardware
//
#include <hip/hip_runtime.h>
#include <hip/hip_bf16.h>

#define B_ 4
#define N_ 4096
#define D_ 1024
#define H_ 16
#define PROJ_ 256
#define DH_ 64   // head dim

typedef unsigned short u16;
typedef __attribute__((ext_vector_type(2)))  float  v2f;
typedef __attribute__((ext_vector_type(8)))  float  v8f;
typedef __attribute__((ext_vector_type(16))) __bf16 v16bf;

union BFrag4 { uint4 q[2]; unsigned u[8]; v16bf v; };

// ---------------- fp32 WMMA (16x16x4) ----------------
__device__ __forceinline__ v8f wmma4(v2f a, v2f b, v8f c) {
  return __builtin_amdgcn_wmma_f32_16x16x4_f32(false, a, false, b, (short)0, c,
                                               false, false);
}
// ---------------- bf16 WMMA (16x16x32) ----------------
__device__ __forceinline__ v8f wmma32(v16bf a, v16bf b, v8f c) {
  return __builtin_amdgcn_wmma_f32_16x16x32_bf16(false, a, false, b, (short)0, c,
                                                 false, false);
}
// 3-term split-bf16 MAC: C += Ah*Bh + Ah*Bl + Al*Bh   (err ~2^-18 rel)
__device__ __forceinline__ v8f mac3(v16bf ah, v16bf al, v16bf bh, v16bf bl, v8f c) {
  c = wmma32(ah, bh, c);
  c = wmma32(ah, bl, c);
  c = wmma32(al, bh, c);
  return c;
}

// ---------------- hi/lo split store into separate bf16 planes ----------------
__device__ __forceinline__ void store_split(u16* __restrict__ hi,
                                            u16* __restrict__ lo,
                                            size_t idx, float x) {
  __bf16 h = (__bf16)x;
  hi[idx] = __builtin_bit_cast(unsigned short, h);
  lo[idx] = __builtin_bit_cast(unsigned short, (__bf16)(x - (float)h));
}
// split two f32 into one packed-bf16-pair VGPR for hi and lo
__device__ __forceinline__ void split2(float x0, float x1, unsigned& h, unsigned& l) {
  __bf16 h0 = (__bf16)x0, h1 = (__bf16)x1;
  __bf16 l0 = (__bf16)(x0 - (float)h0), l1 = (__bf16)(x1 - (float)h1);
  h = ((unsigned)__builtin_bit_cast(unsigned short, h1) << 16) |
      __builtin_bit_cast(unsigned short, h0);
  l = ((unsigned)__builtin_bit_cast(unsigned short, l1) << 16) |
      __builtin_bit_cast(unsigned short, l0);
}

// ---------------- fragment loaders (K contiguous in memory) ----------------
// A 16x32: lane m = lane%16, K = {8h+0..7} U {16+8h+0..7}.  p -> &plane[m][k0]
__device__ __forceinline__ v16bf load_afrag(const u16* p, int half) {
  const uint4* q = (const uint4*)(p + 8 * half);
  BFrag4 f;
  f.q[0] = q[0];   // K = 8h .. 8h+7
  f.q[1] = q[2];   // K = 16+8h .. 16+8h+7
  return f.v;
}
// B 32x16: lane col n = lane%16, K = {8h+0..7} U {16+8h+0..7}.  p -> &planeT[col][k0]
__device__ __forceinline__ v16bf load_bfrag(const u16* p, int half) {
  const uint4* q = (const uint4*)(p + 8 * half);
  BFrag4 f;
  f.q[0] = q[0];
  f.q[1] = q[2];
  return f.v;
}
// A fragment from raw f32 (row-contiguous), in-register hi/lo split
__device__ __forceinline__ void load_afrag_f32(const float* p, int half,
                                               v16bf& ah, v16bf& al) {
  const float4* q0 = (const float4*)(p + 8 * half);
  const float4* q1 = (const float4*)(p + 16 + 8 * half);
  float4 x0 = q0[0], x1 = q0[1], y0 = q1[0], y1 = q1[1];
  BFrag4 H, L;
  split2(x0.x, x0.y, H.u[0], L.u[0]);
  split2(x0.z, x0.w, H.u[1], L.u[1]);
  split2(x1.x, x1.y, H.u[2], L.u[2]);
  split2(x1.z, x1.w, H.u[3], L.u[3]);
  split2(y0.x, y0.y, H.u[4], L.u[4]);
  split2(y0.z, y0.w, H.u[5], L.u[5]);
  split2(y1.x, y1.y, H.u[6], L.u[6]);
  split2(y1.z, y1.w, H.u[7], L.u[7]);
  ah = H.v; al = L.v;
}

// ---------------------------------------------------------------------------
// Kernel 0: pack W_out into hi/lo planes (reused 256x downstream)
// ---------------------------------------------------------------------------
__global__ __launch_bounds__(256) void packW_kernel(const float* __restrict__ W,
                                                    u16* __restrict__ Wh,
                                                    u16* __restrict__ Wl) {
  const int i = blockIdx.x * 256 + threadIdx.x;
  store_split(Wh, Wl, i, W[i]);
}

// ---------------------------------------------------------------------------
// Kernel 1: k_proj[b,h,k,d] = scale * sum_n K[b,n,h*64+d] * E[h,n,k]
// f32 WMMA (K=4096 reduction in full fp32).  Outputs:
//   kp   planes [bh][k][d]   (k-rows, d contiguous)
//   vpT  planes [bh][d][k]   (transposed: k contiguous, for attn@V B-frags)
// grid = (PROJ/64, B*H), block = 128
// ---------------------------------------------------------------------------
__global__ __launch_bounds__(128) void proj_kernel(
    const float* __restrict__ Kin, const float* __restrict__ Vin,
    const float* __restrict__ E, const float* __restrict__ F,
    const float* __restrict__ valence, const float* __restrict__ vscale,
    u16* __restrict__ kpH, u16* __restrict__ kpL,
    u16* __restrict__ vpH, u16* __restrict__ vpL) {
  const int tid  = threadIdx.x;
  const int wave = tid >> 5;
  const int lane = tid & 31;
  const int half = lane >> 4;
  const int mrow = lane & 15;

  const int bh = blockIdx.y;
  const int b  = bh / H_;
  const int h  = bh % H_;
  const int m0 = blockIdx.x * 64 + wave * 16;  // k-row tile of PROJ

  const float scale = 1.0f + vscale[0] *
      0.25f * (valence[0] + valence[1] + valence[2] + valence[3]);

  const float* Eh = E + (size_t)h * N_ * PROJ_;
  const float* Fh = F + (size_t)h * N_ * PROJ_;
  const float* Kb = Kin + (size_t)b * N_ * D_ + h * DH_;
  const float* Vb = Vin + (size_t)b * N_ * D_ + h * DH_;

  v8f ck[4], cv[4];
  for (int j = 0; j < 4; ++j) { ck[j] = (v8f)0.0f; cv[j] = (v8f)0.0f; }

  for (int n0 = 0; n0 < N_; n0 += 4) {
    const int kA = n0 + 2 * half;
    v2f aE, aF;
    aE.x = Eh[(size_t)(kA + 0) * PROJ_ + m0 + mrow];
    aE.y = Eh[(size_t)(kA + 1) * PROJ_ + m0 + mrow];
    aF.x = Fh[(size_t)(kA + 0) * PROJ_ + m0 + mrow];
    aF.y = Fh[(size_t)(kA + 1) * PROJ_ + m0 + mrow];
#pragma unroll
    for (int j = 0; j < 4; ++j) {
      const int col = 16 * j + mrow;
      v2f bK, bV;
      bK.x = Kb[(size_t)(kA + 0) * D_ + col];
      bK.y = Kb[(size_t)(kA + 1) * D_ + col];
      bV.x = Vb[(size_t)(kA + 0) * D_ + col];
      bV.y = Vb[(size_t)(kA + 1) * D_ + col];
      ck[j] = wmma4(aE, bK, ck[j]);
      cv[j] = wmma4(aF, bV, cv[j]);
    }
  }

  const size_t base = (size_t)bh * PROJ_ * DH_;
#pragma unroll
  for (int j = 0; j < 4; ++j)
#pragma unroll
    for (int v = 0; v < 8; ++v) {
      const int m = m0 + v + 8 * half;     // k index
      const int c = 16 * j + mrow;         // d index
      store_split(kpH, kpL, base + (size_t)m * DH_ + c, ck[j][v] * scale);
      store_split(vpH, vpL, base + (size_t)c * PROJ_ + m, cv[j][v] * scale);
    }
}

// ---------------------------------------------------------------------------
// Kernel 2: kkt[b,h] = k_proj @ k_proj^T (256x256, K=64), split-bf16 3-term.
// Result is symmetric, stored row-major (rows serve as K-contiguous B-frags).
// grid = (PROJ/16, 1, B*H), block = 128 (wave w: cols 64w..64w+63)
// ---------------------------------------------------------------------------
__global__ __launch_bounds__(128) void kkt_kernel(
    const u16* __restrict__ kpH, const u16* __restrict__ kpL,
    u16* __restrict__ kkH, u16* __restrict__ kkL) {
  const int tid  = threadIdx.x;
  const int wave = tid >> 5;
  const int lane = tid & 31;
  const int half = lane >> 4;
  const int mrow = lane & 15;

  const int bh = blockIdx.z;
  const int m0 = blockIdx.x * 16;
  const int c0 = wave * 64;

  const u16* kh = kpH + (size_t)bh * PROJ_ * DH_;
  const u16* kl = kpL + (size_t)bh * PROJ_ * DH_;
  v8f acc[4];
  for (int j = 0; j < 4; ++j) acc[j] = (v8f)0.0f;

  for (int k0 = 0; k0 < DH_; k0 += 32) {
    v16bf ah = load_afrag(kh + (m0 + mrow) * DH_ + k0, half);
    v16bf al = load_afrag(kl + (m0 + mrow) * DH_ + k0, half);
#pragma unroll
    for (int j = 0; j < 4; ++j) {
      const int col = c0 + 16 * j + mrow;
      v16bf bh_ = load_bfrag(kh + col * DH_ + k0, half);  // B[k][col]=kp[col][k]
      v16bf bl_ = load_bfrag(kl + col * DH_ + k0, half);
      acc[j] = mac3(ah, al, bh_, bl_, acc[j]);
    }
  }

  const size_t base = (size_t)bh * PROJ_ * PROJ_;
#pragma unroll
  for (int j = 0; j < 4; ++j)
#pragma unroll
    for (int v = 0; v < 8; ++v)
      store_split(kkH, kkL,
                  base + (size_t)(m0 + v + 8 * half) * PROJ_ + c0 + 16 * j + mrow,
                  acc[j][v]);
}

// ---------------------------------------------------------------------------
// Row softmax over sS[16][256] using all 128 threads.
// mode 0: write hi/lo bf16 probs into sPh/sPl
// mode 1: add probs * wgt into sAcc
// ---------------------------------------------------------------------------
__device__ __forceinline__ void row_softmax(float* sS, u16* sPh, u16* sPl,
                                            float* sRed, float* sAcc,
                                            int tid, int mode, float wgt) {
  const int row = tid >> 3;   // 0..15
  const int sub = tid & 7;    // 8 chunks of 32
  float* rp = sS + row * 256 + sub * 32;

  float mx = -3.0e38f;
#pragma unroll
  for (int i = 0; i < 32; ++i) mx = fmaxf(mx, rp[i]);
  sRed[tid] = mx;
  __syncthreads();
  if (sub == 0) {
    float mm = sRed[row * 8];
    for (int k = 1; k < 8; ++k) mm = fmaxf(mm, sRed[row * 8 + k]);
    sRed[row * 8] = mm;
  }
  __syncthreads();
  const float rmax = sRed[row * 8];
  __syncthreads();

  float sum = 0.0f;
#pragma unroll
  for (int i = 0; i < 32; ++i) {
    const float e = __expf(rp[i] - rmax);
    rp[i] = e;
    sum += e;
  }
  sRed[tid] = sum;
  __syncthreads();
  if (sub == 0) {
    float ss = sRed[row * 8];
    for (int k = 1; k < 8; ++k) ss += sRed[row * 8 + k];
    sRed[row * 8] = ss;
  }
  __syncthreads();
  const float inv = 1.0f / sRed[row * 8];
  __syncthreads();

  if (mode == 0) {
    const int off = row * 256 + sub * 32;
#pragma unroll
    for (int i = 0; i < 32; ++i)
      store_split(sPh, sPl, off + i, rp[i] * inv);
  } else {
    float* ap = sAcc + row * 256 + sub * 32;
#pragma unroll
    for (int i = 0; i < 32; ++i) ap[i] += rp[i] * inv * wgt;
  }
  __syncthreads();
}

// ---------------------------------------------------------------------------
// Kernel 3: fused attention per (b, 16-row n-tile), looping all 16 heads.
//   scores = Q_tile @ k_proj^T (split-bf16) -> softmax -> bf16 probs in LDS
//   ctx    = probs @ v_proj  (stored as hi/lo planes for the out-projection)
//   approx = mean_h softmax(probs @ kkt)
// grid = (N/16, B), block = 128 (4 waves)
// ---------------------------------------------------------------------------
__global__ __launch_bounds__(128) void attn_kernel(
    const float* __restrict__ Q,
    const u16* __restrict__ kpH, const u16* __restrict__ kpL,
    const u16* __restrict__ vpH, const u16* __restrict__ vpL,
    const u16* __restrict__ kkH, const u16* __restrict__ kkL,
    u16* __restrict__ ctxH, u16* __restrict__ ctxL,
    float* __restrict__ approx) {
  __shared__ float sS[16 * 256];
  __shared__ u16   sPh[16 * 256];
  __shared__ u16   sPl[16 * 256];
  __shared__ float sAcc[16 * 256];
  __shared__ float sRed[128];

  const int tid  = threadIdx.x;
  const int wave = tid >> 5;
  const int lane = tid & 31;
  const int half = lane >> 4;
  const int mrow = lane & 15;

  const int n0 = blockIdx.x * 16;
  const int b  = blockIdx.y;

#pragma unroll
  for (int i = 0; i < 32; ++i) sAcc[tid * 32 + i] = 0.0f;
  __syncthreads();

  for (int h = 0; h < H_; ++h) {
    const int bh = b * H_ + h;
    const u16* kh = kpH + (size_t)bh * PROJ_ * DH_;
    const u16* kl = kpL + (size_t)bh * PROJ_ * DH_;
    const u16* vh = vpH + (size_t)bh * PROJ_ * DH_;   // transposed [d][k]
    const u16* vl = vpL + (size_t)bh * PROJ_ * DH_;
    const u16* gh = kkH + (size_t)bh * PROJ_ * PROJ_;
    const u16* gl = kkL + (size_t)bh * PROJ_ * PROJ_;
    const float* Qb = Q + (size_t)b * N_ * D_ + (size_t)n0 * D_ + h * DH_;

    // ---- scores: wave w -> columns [64w, 64w+64) of PROJ ----
    v8f sc[4];
    for (int j = 0; j < 4; ++j) sc[j] = (v8f)0.0f;
    for (int d0 = 0; d0 < DH_; d0 += 32) {
      v16bf ah, al;
      load_afrag_f32(Qb + mrow * D_ + d0, half, ah, al);   // A[m][d] = Q
#pragma unroll
      for (int j = 0; j < 4; ++j) {
        const int col = 64 * wave + 16 * j + mrow;
        v16bf bh_ = load_bfrag(kh + col * DH_ + d0, half); // B[d][col]=kp[col][d]
        v16bf bl_ = load_bfrag(kl + col * DH_ + d0, half);
        sc[j] = mac3(ah, al, bh_, bl_, sc[j]);
      }
    }
#pragma unroll
    for (int j = 0; j < 4; ++j)
#pragma unroll
      for (int v = 0; v < 8; ++v)
        sS[(v + 8 * half) * 256 + 64 * wave + 16 * j + mrow] = sc[j][v];
    __syncthreads();

    // ---- softmax -> bf16 hi/lo probs in LDS ----
    row_softmax(sS, sPh, sPl, sRed, sAcc, tid, /*mode=*/0, 0.0f);

    // ---- out tile: wave w -> d columns [16w, 16w+16) ----
    v8f oc = (v8f)0.0f;
    for (int k0 = 0; k0 < PROJ_; k0 += 32) {
      v16bf ah = load_afrag(sPh + mrow * 256 + k0, half);  // probs (LDS)
      v16bf al = load_afrag(sPl + mrow * 256 + k0, half);
      const int dcol = 16 * wave + mrow;
      v16bf bh_ = load_bfrag(vh + dcol * PROJ_ + k0, half); // B[k][d]=vpT[d][k]
      v16bf bl_ = load_bfrag(vl + dcol * PROJ_ + k0, half);
      oc = mac3(ah, al, bh_, bl_, oc);
    }
#pragma unroll
    for (int v = 0; v < 8; ++v)
      store_split(ctxH, ctxL,
                  (size_t)(b * N_ + n0 + v + 8 * half) * D_ + h * DH_ +
                      16 * wave + mrow,
                  oc[v]);

    // ---- approx logits: wave w -> columns [64w, 64w+64) ----
    v8f lg[4];
    for (int j = 0; j < 4; ++j) lg[j] = (v8f)0.0f;
    for (int k0 = 0; k0 < PROJ_; k0 += 32) {
      v16bf ah = load_afrag(sPh + mrow * 256 + k0, half);
      v16bf al = load_afrag(sPl + mrow * 256 + k0, half);
#pragma unroll
      for (int j = 0; j < 4; ++j) {
        const int col = 64 * wave + 16 * j + mrow;
        // kkt symmetric: B[k][col] = kk[col][k]  (k contiguous)
        v16bf bh_ = load_bfrag(gh + (size_t)col * PROJ_ + k0, half);
        v16bf bl_ = load_bfrag(gl + (size_t)col * PROJ_ + k0, half);
        lg[j] = mac3(ah, al, bh_, bl_, lg[j]);
      }
    }
    __syncthreads();   // all reads of sP/sS done before overwriting sS
#pragma unroll
    for (int j = 0; j < 4; ++j)
#pragma unroll
      for (int v = 0; v < 8; ++v)
        sS[(v + 8 * half) * 256 + 64 * wave + 16 * j + mrow] = lg[j][v];
    __syncthreads();

    // ---- softmax + accumulate mean over heads ----
    row_softmax(sS, sPh, sPl, sRed, sAcc, tid, /*mode=*/1, 1.0f / (float)H_);
  }

  // write approx tile
#pragma unroll
  for (int i = 0; i < 32; ++i) {
    const int idx = tid * 32 + i;      // 0..4095
    const int m = idx >> 8;
    const int l = idx & 255;
    approx[(size_t)(b * N_ + n0 + m) * PROJ_ + l] = sAcc[idx];
  }
}

// ---------------------------------------------------------------------------
// Kernel 4: out = ctx(16384x1024) @ W_out^T + b_out, split-bf16 3-term
// grid = (D/64, B*N/64), block = 128; wave w: rows [16w,16w+16), 64 cols
// ---------------------------------------------------------------------------
__global__ __launch_bounds__(128) void outproj_kernel(
    const u16* __restrict__ ctxH, const u16* __restrict__ ctxL,
    const u16* __restrict__ Wh, const u16* __restrict__ Wl,
    const float* __restrict__ bout, float* __restrict__ out) {
  const int tid  = threadIdx.x;
  const int wave = tid >> 5;
  const int lane = tid & 31;
  const int half = lane >> 4;
  const int mrow = lane & 15;

  const int m0 = blockIdx.y * 64 + 16 * wave;
  const int j0 = blockIdx.x * 64;

  v8f acc[4];
  for (int j = 0; j < 4; ++j) acc[j] = (v8f)0.0f;

  for (int k0 = 0; k0 < D_; k0 += 32) {
    v16bf ah = load_afrag(ctxH + (size_t)(m0 + mrow) * D_ + k0, half);
    v16bf al = load_afrag(ctxL + (size_t)(m0 + mrow) * D_ + k0, half);
#pragma unroll
    for (int j = 0; j < 4; ++j) {
      const int col = j0 + 16 * j + mrow;
      v16bf bh_ = load_bfrag(Wh + (size_t)col * D_ + k0, half);  // B[k][col]=W[col][k]
      v16bf bl_ = load_bfrag(Wl + (size_t)col * D_ + k0, half);
      acc[j] = mac3(ah, al, bh_, bl_, acc[j]);
    }
  }
#pragma unroll
  for (int j = 0; j < 4; ++j)
#pragma unroll
    for (int v = 0; v < 8; ++v) {
      const int col = j0 + 16 * j + mrow;
      out[(size_t)(m0 + v + 8 * half) * D_ + col] = acc[j][v] + bout[col];
    }
}

// ---------------------------------------------------------------------------
extern "C" void kernel_launch(void* const* d_in, const int* in_sizes, int n_in,
                              void* d_out, int out_size, void* d_ws, size_t ws_size,
                              hipStream_t stream) {
  const float* Q   = (const float*)d_in[0];
  const float* Kin = (const float*)d_in[1];
  const float* V   = (const float*)d_in[2];
  const float* val = (const float*)d_in[3];
  const float* E   = (const float*)d_in[4];
  const float* F   = (const float*)d_in[5];
  const float* vs  = (const float*)d_in[6];
  const float* W   = (const float*)d_in[7];
  const float* bo  = (const float*)d_in[8];

  float* out    = (float*)d_out;                       // (B,N,D)
  float* approx = out + (size_t)B_ * N_ * D_;          // (B,N,PROJ)

  const size_t SZ_P  = (size_t)B_ * H_ * PROJ_ * DH_;    // 1 Mi elems
  const size_t SZ_KK = (size_t)B_ * H_ * PROJ_ * PROJ_;  // 4 Mi elems
  const size_t SZ_C  = (size_t)B_ * N_ * D_;             // 16 Mi elems
  const size_t SZ_W  = (size_t)D_ * D_;                  // 1 Mi elems

  u16* kpH  = (u16*)d_ws;
  u16* kpL  = kpH + SZ_P;
  u16* vpH  = kpL + SZ_P;
  u16* vpL  = vpH + SZ_P;
  u16* kkH  = vpL + SZ_P;
  u16* kkL  = kkH + SZ_KK;
  u16* ctxH = kkL + SZ_KK;
  u16* ctxL = ctxH + SZ_C;
  u16* Wh   = ctxL + SZ_C;
  u16* Wl   = Wh + SZ_W;

  packW_kernel<<<dim3((D_ * D_) / 256), 256, 0, stream>>>(W, Wh, Wl);
  proj_kernel<<<dim3(PROJ_ / 64, B_ * H_), 128, 0, stream>>>(
      Kin, V, E, F, val, vs, kpH, kpL, vpH, vpL);
  kkt_kernel<<<dim3(PROJ_ / 16, 1, B_ * H_), 128, 0, stream>>>(
      kpH, kpL, kkH, kkL);
  attn_kernel<<<dim3(N_ / 16, B_), 128, 0, stream>>>(
      Q, kpH, kpL, vpH, vpL, kkH, kkL, ctxH, ctxL, approx);
  outproj_kernel<<<dim3(D_ / 64, (B_ * N_) / 64), 128, 0, stream>>>(
      ctxH, ctxL, Wh, Wl, bo, out);
}
